// GRUFeatureExtractor_81381040325169
// MI455X (gfx1250) — compile-verified, matrix operand order checked by reading, not run on hardware
//
#include <hip/hip_runtime.h>
#include <hip/hip_bf16.h>
#include <math.h>

#define B_TOT 4096
#define T_LEN 256
#define DIN 128
#define H 64
#define G3 192
#define ROWS_PER_WAVE 16
#define WAVES_PER_WG 2
#define ROWS_PER_WG (ROWS_PER_WAVE * WAVES_PER_WG)
#define BLOCK_T (32 * WAVES_PER_WG)

typedef __attribute__((ext_vector_type(16))) __bf16 v16bf;
typedef __attribute__((ext_vector_type(8)))  __bf16 v8bf;
typedef __attribute__((ext_vector_type(8)))  float  v8f;

// Native f32 -> bf16 (v_cvt_pk_bf16_f32 on gfx1250).
__device__ __forceinline__ __bf16 f2bf(float f) { return (__bf16)f; }

__device__ __forceinline__ v8f splat8(float v) {
  v8f r;
  for (int i = 0; i < 8; ++i) r[i] = v;
  return r;
}

__device__ __forceinline__ v8f wmma_bf(v16bf a, v16bf b, v8f c) {
  return __builtin_amdgcn_wmma_f32_16x16x32_bf16(false, a, false, b, (short)0, c,
                                                 false, false);
}

// gfx1250 V_TANH_F32 transcendental (co-executes with WMMA per ISA 8.4).
__device__ __forceinline__ float fast_tanh(float x) {
#if __has_builtin(__builtin_amdgcn_tanhf)
  return __builtin_amdgcn_tanhf(x);
#elif __has_builtin(__builtin_amdgcn_tanh_f32)
  return __builtin_amdgcn_tanh_f32(x);
#else
  float r;
  asm volatile("v_tanh_f32 %0, %1\n\tv_nop" : "=v"(r) : "v"(x));
  return r;
#endif
}

// sigmoid(x) = 0.5*tanh(x/2) + 0.5  -> one trans op + FMAs
__device__ __forceinline__ float sigm(float x) {
  return __builtin_fmaf(0.5f, fast_tanh(0.5f * x), 0.5f);
}

// A/B fragment gather from bf16 LDS (row-major, stride in elements).
// Caller pre-applies row*stride + k0 + kh; layout: 8 elems at p, 8 at p+16.
__device__ __forceinline__ v16bf frag_bf(const __bf16* p) {
  v8bf lo = *(const v8bf*)(p);
  v8bf hi = *(const v8bf*)(p + 16);
  return __builtin_shufflevector(lo, hi, 0, 1, 2, 3, 4, 5, 6, 7,
                                 8, 9, 10, 11, 12, 13, 14, 15);
}

// A fragment built directly from global f32 (x), converted to bf16 in regs.
__device__ __forceinline__ v16bf frag_f32(const float* p) {
  float4 a = *(const float4*)(p);
  float4 b = *(const float4*)(p + 4);
  float4 c = *(const float4*)(p + 16);
  float4 d = *(const float4*)(p + 20);
  v16bf r;
  r[0]  = f2bf(a.x); r[1]  = f2bf(a.y); r[2]  = f2bf(a.z); r[3]  = f2bf(a.w);
  r[4]  = f2bf(b.x); r[5]  = f2bf(b.y); r[6]  = f2bf(b.z); r[7]  = f2bf(b.w);
  r[8]  = f2bf(c.x); r[9]  = f2bf(c.y); r[10] = f2bf(c.z); r[11] = f2bf(c.w);
  r[12] = f2bf(d.x); r[13] = f2bf(d.y); r[14] = f2bf(d.z); r[15] = f2bf(d.w);
  return r;
}

// LDS layout (bytes)
#define OFF_WIH0 0                       // 192*128 bf16 = 49152
#define OFF_WHH0 49152                   // 192*64 bf16  = 24576
#define OFF_WIH1 73728
#define OFF_WHH1 98304
#define OFF_WFC  122880                  // 64 f32 = 256
#define OFF_H1   123136                  // 2 waves * 16*64 bf16 = 4096
#define OFF_H2   127232
#define OFF_H2F  131328                  // 2 waves * 16*64 f32 = 8192
#define SMEM_BYTES 139520

__global__ void __launch_bounds__(BLOCK_T)
gru2_fused_kernel(const float* __restrict__ x,
                  const float* __restrict__ Wih0, const float* __restrict__ Whh0,
                  const float* __restrict__ bih0, const float* __restrict__ bhh0,
                  const float* __restrict__ Wih1, const float* __restrict__ Whh1,
                  const float* __restrict__ bih1, const float* __restrict__ bhh1,
                  const float* __restrict__ Wfc,  const float* __restrict__ bfc,
                  float* __restrict__ out) {
  extern __shared__ char smem[];
  __bf16* wih0 = (__bf16*)(smem + OFF_WIH0);
  __bf16* whh0 = (__bf16*)(smem + OFF_WHH0);
  __bf16* wih1 = (__bf16*)(smem + OFF_WIH1);
  __bf16* whh1 = (__bf16*)(smem + OFF_WHH1);
  float*  wfc  = (float*)(smem + OFF_WFC);
  __bf16* h1b  = (__bf16*)(smem + OFF_H1);
  __bf16* h2b  = (__bf16*)(smem + OFF_H2);
  float*  h2s  = (float*)(smem + OFF_H2F);

  const int tid = threadIdx.x;

  // ---- one-time weight conversion to LDS-resident bf16 ----
  for (int i = tid; i < G3 * DIN; i += BLOCK_T) wih0[i] = f2bf(Wih0[i]);
  for (int i = tid; i < G3 * H; i += BLOCK_T) {
    whh0[i] = f2bf(Whh0[i]);
    wih1[i] = f2bf(Wih1[i]);
    whh1[i] = f2bf(Whh1[i]);
  }
  if (tid < H) wfc[tid] = Wfc[tid];
  {
    const __bf16 z = (__bf16)0.0f;
    for (int i = tid; i < WAVES_PER_WG * ROWS_PER_WAVE * H; i += BLOCK_T) {
      h1b[i] = z;
      h2b[i] = z;
    }
  }
  __syncthreads();

  const int wave  = tid >> 5;
  const int lane  = tid & 31;
  const int nc    = lane & 15;          // N column / A row within tile
  const int mbase = (lane >> 4) << 3;   // K-half offset (A/B) == M-half base (C)
  __bf16* h1w  = h1b + wave * ROWS_PER_WAVE * H;
  __bf16* h2w  = h2b + wave * ROWS_PER_WAVE * H;
  float*  h2fw = h2s + wave * ROWS_PER_WAVE * H;
  const int b0 = blockIdx.x * ROWS_PER_WG + wave * ROWS_PER_WAVE;

  // ---- hoist all K=64 (recurrent-path) B fragments into registers for the
  // whole T loop: 3 matrices * 12 N-tiles * 2 K-chunks = 72 frags = 576 VGPRs.
  // CDNA5 wave32 has 1024 VGPRs/wave; this removes the LDS round-trip from
  // 72 of the 120 WMMAs per step. W_ih0 (K=128 projection, off the h->h
  // critical path) keeps streaming from LDS.
  v16bf whh0F[4][3][2], wih1F[4][3][2], whh1F[4][3][2];
#pragma unroll
  for (int i = 0; i < 4; ++i)
#pragma unroll
    for (int g = 0; g < 3; ++g)
#pragma unroll
      for (int k = 0; k < 2; ++k) {
        const int c = g * H + i * 16 + nc;
        whh0F[i][g][k] = frag_bf(whh0 + c * H + k * 32 + mbase);
        wih1F[i][g][k] = frag_bf(wih1 + c * H + k * 32 + mbase);
        whh1F[i][g][k] = frag_bf(whh1 + c * H + k * 32 + mbase);
      }

  // ---- per-lane gate biases (r/z pre-summed; n kept split for r*gh_n) ----
  float brs0[4], bzs0[4], bin0[4], bhn0[4];
  float brs1[4], bzs1[4], bin1[4], bhn1[4];
#pragma unroll
  for (int i = 0; i < 4; ++i) {
    int c = i * 16 + nc;
    brs0[i] = bih0[c] + bhh0[c];
    bzs0[i] = bih0[H + c] + bhh0[H + c];
    bin0[i] = bih0[2 * H + c];
    bhn0[i] = bhh0[2 * H + c];
    brs1[i] = bih1[c] + bhh1[c];
    bzs1[i] = bih1[H + c] + bhh1[H + c];
    bin1[i] = bih1[2 * H + c];
    bhn1[i] = bhh1[2 * H + c];
  }

  // Hidden state kept as packed bf16 in registers (C layout), 16 VGPRs/layer.
  v8bf h1R[4], h2R[4];
#pragma unroll
  for (int i = 0; i < 4; ++i)
#pragma unroll
    for (int e = 0; e < 8; ++e) { h1R[i][e] = (__bf16)0.0f; h2R[i][e] = (__bf16)0.0f; }

  const float* xrow = x + (size_t)(b0 + nc) * (T_LEN * DIN);

#pragma unroll 1
  for (int t = 0; t < T_LEN; ++t) {
    // A fragments of x for this step: K = 0..127 in 4 chunks, direct HBM->regs
    const float* xp = xrow + (size_t)t * DIN + mbase;
    v16bf xf[4];
#pragma unroll
    for (int k = 0; k < 4; ++k) xf[k] = frag_f32(xp + k * 32);
    if (t + 1 < T_LEN) __builtin_prefetch(xp + DIN, 0, 0);

    // A fragments of h1_prev (bf16, LDS)
    v16bf h1f[2];
#pragma unroll
    for (int k = 0; k < 2; ++k) h1f[k] = frag_bf(h1w + nc * H + k * 32 + mbase);

    // ---------------- layer 0 ----------------
#pragma unroll
    for (int i = 0; i < 4; ++i) {
      v8f pr = splat8(0.f), pz = splat8(0.f), pn = splat8(0.f);
      v8f hr = splat8(0.f), hz = splat8(0.f), hn = splat8(0.f);
      const int c = i * 16 + nc;
#pragma unroll
      for (int k = 0; k < 4; ++k) {   // projection: B streams from LDS
        const __bf16* wb = wih0 + c * DIN + k * 32 + mbase;
        pr = wmma_bf(xf[k], frag_bf(wb), pr);
        pz = wmma_bf(xf[k], frag_bf(wb + H * DIN), pz);
        pn = wmma_bf(xf[k], frag_bf(wb + 2 * H * DIN), pn);
      }
#pragma unroll
      for (int k = 0; k < 2; ++k) {   // recurrence: B is register-resident
        hr = wmma_bf(h1f[k], whh0F[i][0][k], hr);
        hz = wmma_bf(h1f[k], whh0F[i][1][k], hz);
        hn = wmma_bf(h1f[k], whh0F[i][2][k], hn);
      }
#pragma unroll
      for (int e = 0; e < 8; ++e) {   // biases folded into the epilogue
        float r  = sigm(pr[e] + hr[e] + brs0[i]);
        float z  = sigm(pz[e] + hz[e] + bzs0[i]);
        float nn = fast_tanh(pn[e] + bin0[i] + r * (hn[e] + bhn0[i]));
        float hprev = (float)h1R[i][e];
        __bf16 hb = f2bf((1.f - z) * nn + z * hprev);
        h1R[i][e] = hb;
        h1w[(mbase + e) * H + i * 16 + nc] = hb; // A-layout LDS mirror
      }
    }

    // ---------------- layer 1 (all B operands register-resident) ----------------
    v16bf g1f[2], h2f[2];
#pragma unroll
    for (int k = 0; k < 2; ++k) {
      g1f[k] = frag_bf(h1w + nc * H + k * 32 + mbase); // h1_new
      h2f[k] = frag_bf(h2w + nc * H + k * 32 + mbase); // h2_prev
    }
#pragma unroll
    for (int i = 0; i < 4; ++i) {
      v8f pr = splat8(0.f), pz = splat8(0.f), pn = splat8(0.f);
      v8f hr = splat8(0.f), hz = splat8(0.f), hn = splat8(0.f);
#pragma unroll
      for (int k = 0; k < 2; ++k) {
        pr = wmma_bf(g1f[k], wih1F[i][0][k], pr);
        pz = wmma_bf(g1f[k], wih1F[i][1][k], pz);
        pn = wmma_bf(g1f[k], wih1F[i][2][k], pn);
        hr = wmma_bf(h2f[k], whh1F[i][0][k], hr);
        hz = wmma_bf(h2f[k], whh1F[i][1][k], hz);
        hn = wmma_bf(h2f[k], whh1F[i][2][k], hn);
      }
#pragma unroll
      for (int e = 0; e < 8; ++e) {
        float r  = sigm(pr[e] + hr[e] + brs1[i]);
        float z  = sigm(pz[e] + hz[e] + bzs1[i]);
        float nn = fast_tanh(pn[e] + bin1[i] + r * (hn[e] + bhn1[i]));
        float hprev = (float)h2R[i][e];
        __bf16 hb = f2bf((1.f - z) * nn + z * hprev);
        h2R[i][e] = hb;
        h2w[(mbase + e) * H + i * 16 + nc] = hb;
      }
    }
  }

  // ---------------- FC epilogue: out[b] = h2_last[b] . Wfc + bfc ----------------
#pragma unroll
  for (int i = 0; i < 4; ++i)
#pragma unroll
    for (int e = 0; e < 8; ++e)
      h2fw[(mbase + e) * H + i * 16 + nc] = (float)h2R[i][e];
  __syncthreads();  // cross-lane relayout through LDS

  if (lane < 16) {
    float s = bfc[0];
    const float* hp = h2fw + lane * H;
#pragma unroll
    for (int c2 = 0; c2 < H; ++c2) s += hp[c2] * wfc[c2];
    out[b0 + lane] = s;
  }
}

extern "C" void kernel_launch(void* const* d_in, const int* in_sizes, int n_in,
                              void* d_out, int out_size, void* d_ws, size_t ws_size,
                              hipStream_t stream) {
  const float* x    = (const float*)d_in[0];
  const float* Wih0 = (const float*)d_in[1];
  const float* Whh0 = (const float*)d_in[2];
  const float* bih0 = (const float*)d_in[3];
  const float* bhh0 = (const float*)d_in[4];
  const float* Wih1 = (const float*)d_in[5];
  const float* Whh1 = (const float*)d_in[6];
  const float* bih1 = (const float*)d_in[7];
  const float* bhh1 = (const float*)d_in[8];
  const float* Wfc  = (const float*)d_in[9];
  const float* bfc  = (const float*)d_in[10];
  float* out = (float*)d_out;

  dim3 grid(B_TOT / ROWS_PER_WG);  // 128 workgroups
  dim3 block(BLOCK_T);             // 2 waves of 32
  gru2_fused_kernel<<<grid, block, SMEM_BYTES, stream>>>(
      x, Wih0, Whh0, bih0, bhh0, Wih1, Whh1, bih1, bhh1, Wfc, bfc, out);
}
